// MultiHeadSelfAttention_45810121179347
// MI455X (gfx1250) — matrix operand order repaired and checked by reading, not compiled, not run on hardware
//
#include <hip/hip_runtime.h>
#include <hip/hip_bf16.h>
#include <stdint.h>

#define D_MODEL  1024
#define N_HEADS  16
#define HEAD_DIM 64
#define BATCH    4
#define SEQ      2048
#define M_TOTAL  (BATCH * SEQ)

typedef __bf16 bf16;
typedef __attribute__((ext_vector_type(16))) __bf16   v16bf;
typedef __attribute__((ext_vector_type(8)))  float    v8f;
typedef __attribute__((ext_vector_type(4)))  uint32_t v4u;
typedef __attribute__((ext_vector_type(8)))  int      v8i;
typedef __attribute__((ext_vector_type(4)))  int      v4i;

// ---------------------------------------------------------------------------
// WMMA helpers (CDNA5, wave32): D(16x16,f32) = A(16x32,bf16) x B(32x16,bf16) + C
// ---------------------------------------------------------------------------
__device__ __forceinline__ v8f wmma_bf16(v16bf a, v16bf b, v8f c) {
  return __builtin_amdgcn_wmma_f32_16x16x32_bf16(
      /*neg_a=*/false, a, /*neg_b=*/false, b,
      /*c_mod=*/(short)0, c, /*reuse_a=*/false, /*reuse_b=*/false);
}

// A fragment (16x32 MxK). rowptr -> element K=0 of the row owned by this lane
// (m = lane&15). h = lane>>4. VGPR j<4: K=2j+8h ; j>=4: K=16+2(j-4)+8h.
__device__ __forceinline__ v16bf load_frag_a(const bf16* rowptr, int h) {
  union { v16bf v; uint32_t u[8]; } f;
  const uint32_t* p = (const uint32_t*)rowptr;
#pragma unroll
  for (int j = 0; j < 8; ++j) {
    int kb = (j < 4) ? (2 * j + 8 * h) : (8 + 2 * j + 8 * h);
    f.u[j] = p[kb >> 1];
  }
  return f.v;
}

// B fragment (32x16 KxN). colptr -> element K=0 of the column owned by this
// lane (n = lane&15), column data contiguous in K. VGPR j<4: K=2j+8h ; j>=4: K=16+2(j-4)+8h.
__device__ __forceinline__ v16bf load_frag_b(const bf16* colptr, int h) {
  union { v16bf v; uint32_t u[8]; } f;
  const uint32_t* p = (const uint32_t*)colptr;
#pragma unroll
  for (int j = 0; j < 8; ++j) f.u[j] = p[((j < 4) ? (2 * j + 8 * h) : (8 + 2 * j + 8 * h)) >> 1];
  return f.v;
}

// ---------------------------------------------------------------------------
// Tensor Data Mover: 2D tile (tile_rows x tile_cols, bf16) global -> LDS.
// D# packed per CDNA5 ISA. tensor dims == tile dims
// (no OOB clamping needed), dim0 stride = row_stride_elems.
// LDS receives the tile packed row-major: row r at lds_off + r*tile_cols*2.
// Toolchain uses the 6-arg builtin: (v4u, v8i, v4i, v4i, v8i, i32 cpol).
// ---------------------------------------------------------------------------
__device__ __forceinline__ void tdm_load_tile_2d(uint32_t lds_off, const void* gsrc,
                                                 uint32_t tile_cols, uint32_t tile_rows,
                                                 uint32_t row_stride_elems) {
  uint64_t ga = (uint64_t)(uintptr_t)gsrc;
  v4u g0;
  g0[0] = 1u;                                                 // count=1, user D#
  g0[1] = lds_off;                                            // lds_addr (bytes)
  g0[2] = (uint32_t)ga;                                       // global_addr[31:0]
  g0[3] = (uint32_t)((ga >> 32) & 0x01FFFFFFu) | (2u << 30);  // addr[56:32] | type=2

  v8i g1;
  g1[0] = (int)(1u << 16);                                    // data_size=1 (2B/elem)
  g1[1] = (int)((tile_cols & 0xFFFFu) << 16);                 // tensor_dim0 lo16
  g1[2] = (int)(((tile_cols >> 16) & 0xFFFFu) |               // tensor_dim0 hi16
                ((tile_rows & 0xFFFFu) << 16));               // tensor_dim1 lo16
  g1[3] = (int)(((tile_rows >> 16) & 0xFFFFu) |               // tensor_dim1 hi16
                ((tile_cols & 0xFFFFu) << 16));               // tile_dim0
  g1[4] = (int)(tile_rows & 0xFFFFu);                         // tile_dim1 (dim2=0)
  g1[5] = (int)row_stride_elems;                              // dim0_stride lo32
  g1[6] = 0;                                                  // stride hi | dim1_stride
  g1[7] = 0;
  v4i z4 = {0, 0, 0, 0};                                      // groups 2/3 unused (2D)
  v8i z8 = {0, 0, 0, 0, 0, 0, 0, 0};
  __builtin_amdgcn_tensor_load_to_lds(g0, g1, z4, z4, z8, 0);
}

// ---------------------------------------------------------------------------
// fp32 -> bf16 conversion
// ---------------------------------------------------------------------------
__global__ __launch_bounds__(256) void cvt_f32_bf16_k(const float* __restrict__ src,
                                                      bf16* __restrict__ dst, int n) {
  int i = blockIdx.x * blockDim.x + threadIdx.x;
  if (i < n) dst[i] = (bf16)src[i];
}

// ---------------------------------------------------------------------------
// Fused QKV projection: Y = X @ W^T. grid = (M/128, D/64, 3), block = 256.
// TDM stages a 128x64 X tile (16KB) and 64x64 W tile (8KB) per K-stage;
// each of 8 waves computes a 32x32 tile as 2x2 fragments (8 WMMAs/stage).
// Q,K stored [B,H,S,hd]; V stored transposed [B,H,hd,S].
// ---------------------------------------------------------------------------
#define KSTEP 64

__global__ __launch_bounds__(256) void qkv_gemm_k(
    const bf16* __restrict__ X, const bf16* __restrict__ Wq,
    const bf16* __restrict__ Wk, const bf16* __restrict__ Wv,
    bf16* __restrict__ Q, bf16* __restrict__ K, bf16* __restrict__ Vt) {
  __shared__ alignas(16) bf16 tX[128 * KSTEP];  // 16 KB
  __shared__ alignas(16) bf16 tW[64 * KSTEP];   //  8 KB

  const int which = blockIdx.z;
  const bf16* W = (which == 0) ? Wq : (which == 1) ? Wk : Wv;

  const int wave = threadIdx.x >> 5;
  const int lane = threadIdx.x & 31;
  const int h  = lane >> 4;
  const int ln = lane & 15;
  const int wr = wave >> 1;                   // 0..3
  const int wc = wave & 1;                    // 0..1
  const int row0 = blockIdx.x * 128;          // block row base
  const int col0 = blockIdx.y * 64;           // block col base

  v8f a00 = {}, a01 = {}, a10 = {}, a11 = {};

  const uint32_t ldsX = (uint32_t)(size_t)&tX[0];
  const uint32_t ldsW = (uint32_t)(size_t)&tW[0];

  for (int kk = 0; kk < D_MODEL; kk += KSTEP) {
    if (wave == 0) {
      tdm_load_tile_2d(ldsX, X + (size_t)row0 * D_MODEL + kk, KSTEP, 128, D_MODEL);
      tdm_load_tile_2d(ldsW, W + (size_t)col0 * D_MODEL + kk, KSTEP, 64, D_MODEL);
      __builtin_amdgcn_s_wait_tensorcnt(0);
    }
    __syncthreads();  // tile visible to all 8 waves

#pragma unroll
    for (int ks = 0; ks < KSTEP; ks += 32) {
      v16bf fa0 = load_frag_a(&tX[(size_t)(wr * 32 + ln)      * KSTEP + ks], h);
      v16bf fa1 = load_frag_a(&tX[(size_t)(wr * 32 + 16 + ln) * KSTEP + ks], h);
      v16bf fb0 = load_frag_b(&tW[(size_t)(wc * 32 + ln)      * KSTEP + ks], h);
      v16bf fb1 = load_frag_b(&tW[(size_t)(wc * 32 + 16 + ln) * KSTEP + ks], h);
      a00 = wmma_bf16(fa0, fb0, a00);
      a01 = wmma_bf16(fa0, fb1, a01);
      a10 = wmma_bf16(fa1, fb0, a10);
      a11 = wmma_bf16(fa1, fb1, a11);
    }
    __syncthreads();  // protect tiles before next stage's DMA
  }

  // epilogue: lane owns column n, rows r+8h
  const int wrow0 = row0 + wr * 32;
  const int wcol0 = col0 + wc * 32;
#pragma unroll
  for (int t = 0; t < 4; ++t) {
    v8f acc = (t == 0) ? a00 : (t == 1) ? a01 : (t == 2) ? a10 : a11;
    int mbase = wrow0 + ((t >> 1) ? 16 : 0);
    int n  = wcol0 + ((t & 1) ? 16 : 0) + ln;
    int hh = n >> 6;
    int d  = n & 63;
#pragma unroll
    for (int r = 0; r < 8; ++r) {
      int m = mbase + r + 8 * h;
      int b = m / SEQ, s = m % SEQ;
      bf16 val = (bf16)acc[r];
      if (which == 2) {
        Vt[(((size_t)(b * N_HEADS + hh) * HEAD_DIM + d)) * SEQ + s] = val;
      } else {
        bf16* dst = (which == 0) ? Q : K;
        dst[(((size_t)(b * N_HEADS + hh) * SEQ + s)) * HEAD_DIM + d] = val;
      }
    }
  }
}

// ---------------------------------------------------------------------------
// Causal flash attention. One wave per 16-query tile.
// grid = (SEQ/16, B*H), block = 32.
// Q,K: [B,H,S,hd] bf16 ; Vt: [B,H,hd,S] bf16 ; O: [B,S,D] bf16 (head-merged).
// ---------------------------------------------------------------------------
__global__ __launch_bounds__(32) void flash_attn_k(
    const bf16* __restrict__ Q, const bf16* __restrict__ Km,
    const bf16* __restrict__ Vt, bf16* __restrict__ O) {
  __shared__ bf16 Plds[16 * 32];

  const int lane = threadIdx.x & 31;
  const int h  = lane >> 4;
  const int ln = lane & 15;
  const int bh = blockIdx.y;
  const int b  = bh / N_HEADS;
  const int hh = bh % N_HEADS;
  const int q0 = blockIdx.x * 16;
  const float scale = 0.125f;  // 1/sqrt(64)

  const bf16* qrow = Q + ((size_t)bh * SEQ + q0 + ln) * HEAD_DIM;
  v16bf aq0 = load_frag_a(qrow, h);
  v16bf aq1 = load_frag_a(qrow + 32, h);

  float mi[8], li[8];
#pragma unroll
  for (int r = 0; r < 8; ++r) { mi[r] = -1e30f; li[r] = 0.0f; }
  v8f o0 = {}, o1 = {}, o2 = {}, o3 = {};

  const int kend = q0 + 16;
  for (int j0 = 0; j0 < kend; j0 += 32) {
    v8f s0 = {}, s1 = {};
    {
      const bf16* kc0 = Km + ((size_t)bh * SEQ + j0 + ln) * HEAD_DIM;
      const bf16* kc1 = Km + ((size_t)bh * SEQ + j0 + 16 + ln) * HEAD_DIM;
      s0 = wmma_bf16(aq0, load_frag_b(kc0,      h), s0);
      s0 = wmma_bf16(aq1, load_frag_b(kc0 + 32, h), s0);
      s1 = wmma_bf16(aq0, load_frag_b(kc1,      h), s1);
      s1 = wmma_bf16(aq1, load_frag_b(kc1 + 32, h), s1);
    }

    float rowmax[8];
#pragma unroll
    for (int r = 0; r < 8; ++r) {
      int m = q0 + r + 8 * h;
      float v0 = s0[r] * scale; if (j0 + ln      > m) v0 = -1e30f;
      float v1 = s1[r] * scale; if (j0 + 16 + ln > m) v1 = -1e30f;
      s0[r] = v0; s1[r] = v1;
      rowmax[r] = fmaxf(v0, v1);
    }
#pragma unroll
    for (int d = 8; d >= 1; d >>= 1) {
#pragma unroll
      for (int r = 0; r < 8; ++r)
        rowmax[r] = fmaxf(rowmax[r], __shfl_xor(rowmax[r], d, 32));
    }

    float alpha[8], rowsum[8];
#pragma unroll
    for (int r = 0; r < 8; ++r) {
      float mnew = fmaxf(mi[r], rowmax[r]);
      alpha[r] = __expf(mi[r] - mnew);
      mi[r] = mnew;
      float p0 = __expf(s0[r] - mnew);
      float p1 = __expf(s1[r] - mnew);
      rowsum[r] = p0 + p1;
      Plds[(r + 8 * h) * 32 + ln]      = (bf16)p0;
      Plds[(r + 8 * h) * 32 + 16 + ln] = (bf16)p1;
    }
#pragma unroll
    for (int d = 8; d >= 1; d >>= 1) {
#pragma unroll
      for (int r = 0; r < 8; ++r)
        rowsum[r] += __shfl_xor(rowsum[r], d, 32);
    }
#pragma unroll
    for (int r = 0; r < 8; ++r) {
      li[r] = li[r] * alpha[r] + rowsum[r];
      o0[r] *= alpha[r]; o1[r] *= alpha[r];
      o2[r] *= alpha[r]; o3[r] *= alpha[r];
    }

    __syncthreads();  // P: C-layout stores -> A-layout loads via LDS

    v16bf ap = load_frag_a(&Plds[ln * 32], h);
    const bf16* vc = Vt + ((size_t)bh * HEAD_DIM + ln) * SEQ + j0;
    o0 = wmma_bf16(ap, load_frag_b(vc,            h), o0);
    o1 = wmma_bf16(ap, load_frag_b(vc + 16 * SEQ, h), o1);
    o2 = wmma_bf16(ap, load_frag_b(vc + 32 * SEQ, h), o2);
    o3 = wmma_bf16(ap, load_frag_b(vc + 48 * SEQ, h), o3);

    __syncthreads();  // protect Plds before next iteration
  }

#pragma unroll
  for (int r = 0; r < 8; ++r) {
    int s = q0 + r + 8 * h;
    float inv = 1.0f / li[r];
    size_t base = ((size_t)b * SEQ + s) * D_MODEL + (size_t)hh * HEAD_DIM;
    O[base +  0 + ln] = (bf16)(o0[r] * inv);
    O[base + 16 + ln] = (bf16)(o1[r] * inv);
    O[base + 32 + ln] = (bf16)(o2[r] * inv);
    O[base + 48 + ln] = (bf16)(o3[r] * inv);
  }
}

// ---------------------------------------------------------------------------
// Output projection: out = Obf @ Wo^T, fp32 result. TDM-staged like qkv_gemm.
// grid = (M/128, D/64), block = 256.
// ---------------------------------------------------------------------------
__global__ __launch_bounds__(256) void out_proj_k(
    const bf16* __restrict__ X, const bf16* __restrict__ W,
    float* __restrict__ Y) {
  __shared__ alignas(16) bf16 tX[128 * KSTEP];
  __shared__ alignas(16) bf16 tW[64 * KSTEP];

  const int wave = threadIdx.x >> 5;
  const int lane = threadIdx.x & 31;
  const int h  = lane >> 4;
  const int ln = lane & 15;
  const int wr = wave >> 1;
  const int wc = wave & 1;
  const int row0 = blockIdx.x * 128;
  const int col0 = blockIdx.y * 64;

  v8f a00 = {}, a01 = {}, a10 = {}, a11 = {};

  const uint32_t ldsX = (uint32_t)(size_t)&tX[0];
  const uint32_t ldsW = (uint32_t)(size_t)&tW[0];

  for (int kk = 0; kk < D_MODEL; kk += KSTEP) {
    if (wave == 0) {
      tdm_load_tile_2d(ldsX, X + (size_t)row0 * D_MODEL + kk, KSTEP, 128, D_MODEL);
      tdm_load_tile_2d(ldsW, W + (size_t)col0 * D_MODEL + kk, KSTEP, 64, D_MODEL);
      __builtin_amdgcn_s_wait_tensorcnt(0);
    }
    __syncthreads();

#pragma unroll
    for (int ks = 0; ks < KSTEP; ks += 32) {
      v16bf fa0 = load_frag_a(&tX[(size_t)(wr * 32 + ln)      * KSTEP + ks], h);
      v16bf fa1 = load_frag_a(&tX[(size_t)(wr * 32 + 16 + ln) * KSTEP + ks], h);
      v16bf fb0 = load_frag_b(&tW[(size_t)(wc * 32 + ln)      * KSTEP + ks], h);
      v16bf fb1 = load_frag_b(&tW[(size_t)(wc * 32 + 16 + ln) * KSTEP + ks], h);
      a00 = wmma_bf16(fa0, fb0, a00);
      a01 = wmma_bf16(fa0, fb1, a01);
      a10 = wmma_bf16(fa1, fb0, a10);
      a11 = wmma_bf16(fa1, fb1, a11);
    }
    __syncthreads();
  }

  const int wrow0 = row0 + wr * 32;
  const int wcol0 = col0 + wc * 32;
#pragma unroll
  for (int t = 0; t < 4; ++t) {
    v8f acc = (t == 0) ? a00 : (t == 1) ? a01 : (t == 2) ? a10 : a11;
    int mbase = wrow0 + ((t >> 1) ? 16 : 0);
    int n = wcol0 + ((t & 1) ? 16 : 0) + ln;
#pragma unroll
    for (int r = 0; r < 8; ++r) {
      int m = mbase + r + 8 * h;
      Y[(size_t)m * D_MODEL + n] = acc[r];
    }
  }
}

// ---------------------------------------------------------------------------
// Host-side launch
// ---------------------------------------------------------------------------
extern "C" void kernel_launch(void* const* d_in, const int* in_sizes, int n_in,
                              void* d_out, int out_size, void* d_ws, size_t ws_size,
                              hipStream_t stream) {
  const float* x  = (const float*)d_in[0];
  const float* wq = (const float*)d_in[1];
  const float* wk = (const float*)d_in[2];
  const float* wv = (const float*)d_in[3];
  const float* wo = (const float*)d_in[4];
  float* out = (float*)d_out;

  char* ws = (char*)d_ws;
  const size_t MB = 1024ull * 1024ull;
  bf16* Xbf = (bf16*)(ws);                    // 16 MB  : [8192,1024]
  bf16* Wqb = (bf16*)(ws + 16 * MB);          //  2 MB  : [1024,1024]
  bf16* Wkb = (bf16*)(ws + 18 * MB);
  bf16* Wvb = (bf16*)(ws + 20 * MB);
  bf16* Wob = (bf16*)(ws + 22 * MB);
  bf16* Qb  = (bf16*)(ws + 24 * MB);          // 16 MB  : [B,H,S,hd]
  bf16* Kb  = (bf16*)(ws + 40 * MB);          // 16 MB  : [B,H,S,hd]
  bf16* Vtb = (bf16*)(ws + 56 * MB);          // 16 MB  : [B,H,hd,S]
  bf16* Ob  = (bf16*)(ws + 72 * MB);          // 16 MB  : [B,S,D]

  {
    int n = M_TOTAL * D_MODEL;
    cvt_f32_bf16_k<<<n / 256, 256, 0, stream>>>(x, Xbf, n);
    int nw = D_MODEL * D_MODEL;
    cvt_f32_bf16_k<<<nw / 256, 256, 0, stream>>>(wq, Wqb, nw);
    cvt_f32_bf16_k<<<nw / 256, 256, 0, stream>>>(wk, Wkb, nw);
    cvt_f32_bf16_k<<<nw / 256, 256, 0, stream>>>(wv, Wvb, nw);
    cvt_f32_bf16_k<<<nw / 256, 256, 0, stream>>>(wo, Wob, nw);
  }
  {
    dim3 grid(M_TOTAL / 128, D_MODEL / 64, 3);
    qkv_gemm_k<<<grid, 256, 0, stream>>>(Xbf, Wqb, Wkb, Wvb, Qb, Kb, Vtb);
  }
  {
    dim3 grid(SEQ / 16, BATCH * N_HEADS);
    flash_attn_k<<<grid, 32, 0, stream>>>(Qb, Kb, Vtb, Ob);
  }
  {
    dim3 grid(M_TOTAL / 128, D_MODEL / 64);
    out_proj_k<<<grid, 256, 0, stream>>>(Ob, Wob, out);
  }
}
